// ValenceModel_73048803770673
// MI455X (gfx1250) — compile-verified, matrix operand order checked by reading, not run on hardware
//
#include <hip/hip_runtime.h>

// ---------------------------------------------------------------------------
// CDNA5 / gfx1250 valence-readout kernels, v2.
// - node_reps pre-converted once to f16 (25.6MB -> fully L2-resident on the
//   192MB L2): halves gather bytes and makes repeat-gathers HBM-free.
// - MT=2 interaction tiles per workgroup: each W1 B-fragment feeds
//   NPERM*MT WMMAs, halving weight L2 traffic and raising WMMA density.
// - layer1: v_wmma_f32_16x16x32_f16, f32 accum, B shared across perms,
//   perm-sum applied after ReLU (linear), layer2 = tiny padded WMMA GEMM.
// ---------------------------------------------------------------------------

typedef __attribute__((ext_vector_type(16))) _Float16 v16h;
typedef __attribute__((ext_vector_type(8)))  _Float16 v8h;
typedef __attribute__((ext_vector_type(8)))  float    v8f;

#define GROW 136  // padded LDS row stride in halves (272B) -> conflict-free b128 frag reads

__device__ __forceinline__ v16h ld_frag(const _Float16* p) {
  // ISA 7.12.2 16-bit A/B fragment: 16 halves per lane as two contiguous
  // 8-half runs located 16 halves apart (caller supplies lane-half K offset).
  v8h lo = *(const v8h*)(p);
  v8h hi = *(const v8h*)(p + 16);
  return __builtin_shufflevector(lo, hi, 0,1,2,3,4,5,6,7,8,9,10,11,12,13,14,15);
}

// ------------------------- prep kernels (cheap, per-call) ------------------
// node_reps f32 -> f16 table (L2-resident working set afterwards).
__global__ void convert_node_f16(const float4* __restrict__ src,
                                 _Float16* __restrict__ dst, int total4) {
  int i = blockIdx.x * blockDim.x + threadIdx.x;
  if (i < total4) {
    float4 v = src[i];
    _Float16* d = dst + (long)i * 4;
    d[0] = (_Float16)v.x; d[1] = (_Float16)v.y;
    d[2] = (_Float16)v.z; d[3] = (_Float16)v.w;
  }
}

// W1 (f32, [KD][128]) -> W1T (f16, [128][KD]) so B-fragments are contiguous.
__global__ void transpose_w1_f16(const float* __restrict__ w1,
                                 _Float16* __restrict__ w1t, int KD) {
  int i = blockIdx.x * blockDim.x + threadIdx.x;
  int total = KD * 128;
  if (i < total) {
    int r = i >> 7;        // row in KD
    int h = i & 127;       // hidden column
    w1t[(long)h * KD + r] = (_Float16)w1[i];
  }
}

// W2 (f32, [128][DOUT]) -> W2T (f16, [16][128]) zero-padded to 16 columns.
__global__ void pad_w2_f16(const float* __restrict__ w2,
                           _Float16* __restrict__ w2t, int DOUT) {
  int i = blockIdx.x * blockDim.x + threadIdx.x;  // over 16*128
  if (i < 16 * 128) {
    int n = i >> 7;        // output column (padded)
    int h = i & 127;       // hidden row
    w2t[n * 128 + h] = (n < DOUT) ? (_Float16)w2[h * DOUT + n] : (_Float16)0.0f;
  }
}

// ------------------------------ main kernel --------------------------------
// KP: atoms per interaction, NPERM: symmetry perms, DOUT: outputs,
// MT: 16-row M tiles per workgroup.
// permBits: perm[p][i] = (permBits >> (p*8+i*2)) & 3.
template<int KP, int NPERM, int DOUT, int MT>
__global__ __launch_bounds__(256) void valence_kernel(
    const _Float16* __restrict__ nodef16,     // [N][128] f16 (prep output)
    const int*      __restrict__ idx,         // [count][KP] or nullptr (identity)
    const _Float16* __restrict__ w1t,         // [128][KP*128] f16 (transposed)
    const float*    __restrict__ b1,          // [128]
    const _Float16* __restrict__ w2t,         // [16][128] f16 (transposed, padded)
    const float*    __restrict__ b2,          // [DOUT]
    float*          __restrict__ out,         // [count][DOUT]
    int count, unsigned permBits) {
  constexpr int KD = KP * 128;

  __shared__ _Float16 sGath[KP * MT * 16 * GROW];  // gathered f16 reps (padded rows)
  __shared__ _Float16 sHid[MT * 16 * GROW];        // perm-summed ReLU hidden (f16)

  const int tid  = threadIdx.x;
  const int wave = tid >> 5;
  const int lane = tid & 31;
  const int m0   = blockIdx.x * (16 * MT);

  // ---- gather f16 rows straight into LDS (16B per thread, coalesced per row)
  for (int c = tid; c < KP * MT * 16 * 16; c += 256) {  // 16 x 16B chunks per row
    int chunk   = c & 15;
    int rowSlot = c >> 4;                 // (pos*MT + mt)*16 + m
    int m    = rowSlot & 15;
    int mt   = (rowSlot >> 4) % MT;
    int pos  = (rowSlot >> 4) / MT;
    int gm   = m0 + mt * 16 + m;
    if (gm >= count) gm = count - 1;      // clamp tail (store masked later)
    int row = idx ? idx[(long)gm * KP + pos] : gm;
    v8h v = ((const v8h*)(nodef16 + (long)row * 128))[chunk];
    *(v8h*)&sGath[rowSlot * GROW + chunk * 8] = v;
  }
  __syncthreads();

  const int nIn    = lane & 15;           // column-in-tile / M selector
  const int hiHalf = lane >> 4;           // lane half selects K offset (ISA layout)
  const int kOffA  = hiHalf * 8;
  const int nCol   = wave * 16 + nIn;     // this wave's hidden column

  // ---- layer 1: per-(perm, mtile) f32 accumulators; B fragment shared by all
  v8f acc[NPERM][MT];
#pragma unroll
  for (int p = 0; p < NPERM; ++p)
#pragma unroll
    for (int mt = 0; mt < MT; ++mt)
#pragma unroll
      for (int r = 0; r < 8; ++r) acc[p][mt][r] = 0.0f;

#pragma unroll
  for (int i = 0; i < KP; ++i) {
#pragma unroll
    for (int kk = 0; kk < 128; kk += 32) {
      v16h bf = ld_frag(w1t + (long)nCol * KD + i * 128 + kk + kOffA);
#pragma unroll
      for (int p = 0; p < NPERM; ++p) {
        const int a = (int)((permBits >> (p * 8 + i * 2)) & 3u);  // perm[p][i]
#pragma unroll
        for (int mt = 0; mt < MT; ++mt) {
          v16h af = ld_frag(
              &sGath[((a * MT + mt) * 16 + nIn) * GROW + kk + kOffA]);
          acc[p][mt] = __builtin_amdgcn_wmma_f32_16x16x32_f16(
              false, af, false, bf, (short)0, acc[p][mt], false, false);
        }
      }
    }
  }

  // ---- bias + ReLU + perm-sum (linear after ReLU), f16 hidden to LDS
  {
    const float bb = b1[nCol];
#pragma unroll
    for (int mt = 0; mt < MT; ++mt) {
#pragma unroll
      for (int r = 0; r < 8; ++r) {
        float v0 = acc[0][mt][r] + bb;
        float s  = (v0 > 0.0f) ? v0 : 0.0f;
#pragma unroll
        for (int p = 1; p < NPERM; ++p) {
          float v = acc[p][mt][r] + bb;
          s += (v > 0.0f) ? v : 0.0f;
        }
        int m = r + hiHalf * 8;           // C/D layout: VGPR r -> M=r (+8 hi lanes)
        sHid[(mt * 16 + m) * GROW + nCol] = (_Float16)s;
      }
    }
  }
  __syncthreads();

  // ---- layer 2: 16x128 @ 128x16 GEMM, one M-tile per wave (waves 0..MT-1)
  if (wave < MT) {
    v8f acc2;
#pragma unroll
    for (int r = 0; r < 8; ++r) acc2[r] = 0.0f;
#pragma unroll
    for (int kk = 0; kk < 128; kk += 32) {
      v16h af = ld_frag(&sHid[(wave * 16 + nIn) * GROW + kk + kOffA]);
      v16h bf = ld_frag(w2t + nIn * 128 + kk + kOffA);
      acc2 = __builtin_amdgcn_wmma_f32_16x16x32_f16(
          false, af, false, bf, (short)0, acc2, false, false);
    }
    const float b2v = (nIn < DOUT) ? (float)NPERM * b2[nIn] : 0.0f;
#pragma unroll
    for (int r = 0; r < 8; ++r) {
      int m  = r + hiHalf * 8;
      int gm = m0 + wave * 16 + m;
      if (gm < count && nIn < DOUT)
        out[(long)gm * DOUT + nIn] = acc2[r] + b2v;
    }
  }
}

// ------------------------------- host side ---------------------------------
static constexpr unsigned encPerm(int a, int b, int c = 0, int d = 0) {
  return (unsigned)(a | (b << 2) | (c << 4) | (d << 6));
}

extern "C" void kernel_launch(void* const* d_in, const int* in_sizes, int n_in,
                              void* d_out, int out_size, void* d_ws, size_t ws_size,
                              hipStream_t stream) {
  (void)n_in; (void)out_size; (void)ws_size;

  const float* node = (const float*)d_in[0];
  const int*   bidx = (const int*)d_in[1];
  const int*   gidx = (const int*)d_in[2];
  const int*   pidx = (const int*)d_in[3];
  const int*   iidx = (const int*)d_in[4];

  // params in dict order: a,b,g,p,i each (w1,b1,w2,b2) starting at index 5
  const float* W1[5]; const float* B1[5]; const float* W2[5]; const float* B2[5];
  for (int t = 0; t < 5; ++t) {
    W1[t] = (const float*)d_in[5 + 4 * t + 0];
    B1[t] = (const float*)d_in[5 + 4 * t + 1];
    W2[t] = (const float*)d_in[5 + 4 * t + 2];
    B2[t] = (const float*)d_in[5 + 4 * t + 3];
  }
  const int KD[5]   = {128, 256, 384, 512, 512};
  const int DOUT[5] = {2, 2, 2, 6, 6};

  const int N  = in_sizes[0] / 128;
  const int NB = in_sizes[1] / 2;
  const int NA = in_sizes[2] / 3;
  const int NP = in_sizes[3] / 4;
  const int NI = in_sizes[4] / 4;

  // workspace: f16 node table, then f16 transposed W1s, then padded W2s
  _Float16* ws = (_Float16*)d_ws;
  size_t off = 0;
  _Float16* nodeF16 = ws;       off += (size_t)N * 128;
  _Float16* w1t[5]; _Float16* w2t[5];
  for (int t = 0; t < 5; ++t) { w1t[t] = ws + off; off += (size_t)128 * KD[t]; }
  for (int t = 0; t < 5; ++t) { w2t[t] = ws + off; off += 16 * 128; }

  {
    int total4 = N * 32;  // N*128 / 4
    convert_node_f16<<<(total4 + 255) / 256, 256, 0, stream>>>(
        (const float4*)node, nodeF16, total4);
  }
  for (int t = 0; t < 5; ++t) {
    int total = KD[t] * 128;
    transpose_w1_f16<<<(total + 255) / 256, 256, 0, stream>>>(W1[t], w1t[t], KD[t]);
    pad_w2_f16<<<8, 256, 0, stream>>>(W2[t], w2t[t], DOUT[t]);
  }

  float* out    = (float*)d_out;
  float* oAtom  = out;
  float* oBond  = oAtom + (size_t)N * 2;
  float* oAngle = oBond + (size_t)NB * 2;
  float* oProp  = oAngle + (size_t)NA * 2;
  float* oImp   = oProp + (size_t)NP * 6;

  constexpr int MT = 2;
  const int TW = 16 * MT;  // interactions per workgroup

  // atoms: k=1, 1 "perm", dout=2, identity index
  valence_kernel<1, 1, 2, MT><<<(N + TW - 1) / TW, 256, 0, stream>>>(
      nodeF16, nullptr, w1t[0], B1[0], w2t[0], B2[0], oAtom, N, 0u);

  // bonds: perms (0,1),(1,0)
  valence_kernel<2, 2, 2, MT><<<(NB + TW - 1) / TW, 256, 0, stream>>>(
      nodeF16, bidx, w1t[1], B1[1], w2t[1], B2[1], oBond, NB,
      encPerm(0, 1) | (encPerm(1, 0) << 8));

  // angles: perms (0,1,2),(2,1,0)
  valence_kernel<3, 2, 2, MT><<<(NA + TW - 1) / TW, 256, 0, stream>>>(
      nodeF16, gidx, w1t[2], B1[2], w2t[2], B2[2], oAngle, NA,
      encPerm(0, 1, 2) | (encPerm(2, 1, 0) << 8));

  // propers: perms (0,1,2,3),(3,2,1,0)
  valence_kernel<4, 2, 6, MT><<<(NP + TW - 1) / TW, 256, 0, stream>>>(
      nodeF16, pidx, w1t[3], B1[3], w2t[3], B2[3], oProp, NP,
      encPerm(0, 1, 2, 3) | (encPerm(3, 2, 1, 0) << 8));

  // impropers: perms (0,1,2,3),(2,1,3,0),(3,1,0,2)
  valence_kernel<4, 3, 6, MT><<<(NI + TW - 1) / TW, 256, 0, stream>>>(
      nodeF16, iidx, w1t[4], B1[4], w2t[4], B2[4], oImp, NI,
      encPerm(0, 1, 2, 3) | (encPerm(2, 1, 3, 0) << 8) |
          (encPerm(3, 1, 0, 2) << 16));
}
